// Fieldaware_Factorization_Machine_22187801051244
// MI455X (gfx1250) — matrix operand order repaired and checked by reading, not compiled, not run on hardware
//
#include <hip/hip_runtime.h>

// Field-aware FM second-order term on MI455X (gfx1250, wave32).
//
// Roofline: gather traffic ~105 MB (per batch element: 20 features x 320
// floats of field-aware embeddings), compute ~25 MFLOP -> pure gather-
// bandwidth problem (~4.5 us floor at 23.3 TB/s; v is 128 MB, L2-resident).
// Strategy: per 16-batch tile, gather the F x F x K interaction tensor ONCE
// into LDS (f32 -> f16, 204.8 KB of the 320 KB WGP LDS), then evaluate all
// 190 field pairs with v_wmma_f32_16x16x32_f16, packing TWO pairs per WMMA
// (pair1 -> K 0..15, pair2 -> K 16..31) and accumulating every pair into a
// single f32 16x16 accumulator whose diagonal is sum_{i<j} S[i,j,b].

typedef __attribute__((ext_vector_type(16))) _Float16 v16h;
typedef __attribute__((ext_vector_type(8)))  _Float16 v8h;
typedef __attribute__((ext_vector_type(8)))  float    v8f;
typedef __attribute__((ext_vector_type(4)))  float    v4f;

#define FF     20        // fields
#define HH     100000    // hash buckets
#define KK     16        // embedding dim
#define BATCH  4096
#define BTILE  16        // batch elements per workgroup (WMMA M/N dim)
#define NTHR   256       // 8 wave32 waves
#define NWAVES 8
#define NUNITS 95        // 190 pairs (i<j), 2 per WMMA

// Compile-time pair table: unit u covers pairs 2u and 2u+1 of the (i<j)
// enumeration, packed as i1 | j1<<8 | i2<<16 | j2<<24. Uniform scalar loads
// at runtime instead of a lane-divergent decode loop.
struct PairTab { unsigned int p[NUNITS]; };
static constexpr PairTab make_pairs() {
    PairTab t{};
    unsigned char pi[2 * NUNITS] = {}, pj[2 * NUNITS] = {};
    int n = 0;
    for (int i = 0; i < FF; ++i)
        for (int j = i + 1; j < FF; ++j) {
            pi[n] = (unsigned char)i;
            pj[n] = (unsigned char)j;
            ++n;
        }
    for (int u = 0; u < NUNITS; ++u)
        t.p[u] = (unsigned)pi[2 * u] | ((unsigned)pj[2 * u] << 8) |
                 ((unsigned)pi[2 * u + 1] << 16) | ((unsigned)pj[2 * u + 1] << 24);
    return t;
}
__constant__ PairTab g_pairs = make_pairs();

__global__ __launch_bounds__(NTHR)
void ffm_wmma_kernel(const int*   __restrict__ x,    // (F, B)
                     const float* __restrict__ v,    // (F, H, K)
                     const float* __restrict__ w1,   // (H, 1)
                     const float* __restrict__ w0,   // scalar
                     float*       __restrict__ out)  // (B,)
{
    // emb[p][f][b][k] = (f16) v[f, x[p,b], k]
    __shared__ __align__(16) _Float16 s_emb[FF * FF * BTILE * KK]; // 204.8 KB
    __shared__ int   s_x[FF * BTILE];
    __shared__ float s_part[NWAVES][BTILE];   // per-wave diagonal partials

    const int tid  = threadIdx.x;
    const int lane = tid & 31;
    // readfirstlane -> SGPR: guarantees the unit loop below is scalar control
    // flow (full EXEC at every WMMA, no exec-mask waterfalling).
    const int wave = __builtin_amdgcn_readfirstlane(tid) >> 5;
    const int b0   = blockIdx.x * BTILE;

    // ---- stage indices for this batch tile ----
    for (int s = tid; s < FF * BTILE; s += NTHR) {
        int p = s >> 4, b = s & 15;
        s_x[s] = x[p * BATCH + b0 + b];
    }
    __syncthreads();

    // ---- gather v -> LDS, f32 -> f16 (exactly-once gather: 6400 x 64B) ----
    for (int s = tid; s < FF * FF * BTILE; s += NTHR) {     // 25 iters, uniform
        int p = s / (FF * BTILE);
        int r = s - p * (FF * BTILE);
        int f = r >> 4;
        int b = r & 15;
        const v4f* src =
            (const v4f*)(v + ((size_t)f * HH + (size_t)s_x[p * BTILE + b]) * KK);
        v4f a0 = src[0], a1 = src[1], a2 = src[2], a3 = src[3];
        v8h h0, h1;
#pragma unroll
        for (int k = 0; k < 4; ++k) {
            h0[k]     = (_Float16)a0[k];
            h0[4 + k] = (_Float16)a1[k];
            h1[k]     = (_Float16)a2[k];
            h1[4 + k] = (_Float16)a3[k];
        }
        v8h* dst = (v8h*)&s_emb[((p * FF + f) * BTILE + b) * KK];
        dst[0] = h0;
        dst[1] = h1;
    }
    __syncthreads();

    // ---- pairwise terms via WMMA: 95 units of 2 pairs, split over 8 waves --
    // A[m=b, khat]:  khat 0..15 -> v[i1, x[j1,b], k], khat 16..31 -> pair2
    // B[khat, n=b']: khat 0..15 -> v[j1, x[i1,b'],k], khat 16..31 -> pair2
    // diag of the accumulated C is sum over handled pairs of S[i,j,b].
    v8f acc = {};
    const int b  = lane & 15;   // batch element within tile (M and N index)
    const int hi = lane >> 4;   // upper half-wave flag

    for (int u = wave; u < NUNITS; u += NWAVES) {           // scalar loop
        const unsigned pk = g_pairs.p[u];                   // uniform load
        const int i1 =  pk        & 0xff;
        const int j1 = (pk >> 8)  & 0xff;
        const int i2 = (pk >> 16) & 0xff;
        const int j2 = (pk >> 24) & 0xff;

        // A fragment (ISA 16-bit A 16x32 layout):
        //   lanes<16 : halfs0-7 = K0..7   | halfs8-15 = K16..23
        //   lanes>=16: halfs0-7 = K8..15  | halfs8-15 = K24..31
        const v8h aLo = *(const v8h*)&s_emb[((j1 * FF + i1) * BTILE + b) * KK + hi * 8];
        const v8h aHi = *(const v8h*)&s_emb[((j2 * FF + i2) * BTILE + b) * KK + hi * 8];
        // B fragment (same K order as A): lanes<16: halfs0-7 = K0..7 | halfs8-15 = K16..23,
        // lanes>=16: halfs0-7 = K8..15 | halfs8-15 = K24..31; N = lane % 16.
        const v8h bLo = *(const v8h*)&s_emb[((i1 * FF + j1) * BTILE + b) * KK + hi * 8];
        const v8h bHi = *(const v8h*)&s_emb[((i2 * FF + j2) * BTILE + b) * KK + hi * 8];

        v16h A, Bm;
#pragma unroll
        for (int t = 0; t < 8; ++t) {
            A[t]  = aLo[t]; A[8 + t]  = aHi[t];
            Bm[t] = bLo[t]; Bm[8 + t] = bHi[t];
        }
        acc = __builtin_amdgcn_wmma_f32_16x16x32_f16(
                  false, A, false, Bm, (short)0, acc, false, false);
    }

    // ---- extract the diagonal, one slot per wave (deterministic: no atomics)
    // C layout: VGPR r -> (M = r + 8*hi, N = lane%16). Diagonal: M == N,
    // which lives on lanes 0..7 (r = lane) and lanes 24..31 (r = lane - 24).
#pragma unroll
    for (int r = 0; r < 8; ++r) {
        int m = r + (hi << 3);
        if (m == b) s_part[wave][b] = acc[r];               // ds_store_b32
    }
    __syncthreads();

    // ---- fixed-order cross-wave reduction + first-order term + bias ----
    if (tid < BTILE) {
        float f2 = 0.0f;
#pragma unroll
        for (int w = 0; w < NWAVES; ++w) f2 += s_part[w][tid];
        float f1 = w0[0];
#pragma unroll
        for (int p = 0; p < FF; ++p) f1 += w1[s_x[p * BTILE + tid]];
        out[b0 + tid] = f1 + f2;
    }
}

extern "C" void kernel_launch(void* const* d_in, const int* in_sizes, int n_in,
                              void* d_out, int out_size, void* d_ws, size_t ws_size,
                              hipStream_t stream) {
    const int*   x  = (const int*)  d_in[0];   // (F, B) int32
    const float* v  = (const float*)d_in[1];   // (F, H, K) f32
    const float* w1 = (const float*)d_in[2];   // (H, 1) f32
    const float* w0 = (const float*)d_in[3];   // scalar f32
    float*       out = (float*)d_out;          // (B,) f32
    (void)in_sizes; (void)n_in; (void)out_size; (void)d_ws; (void)ws_size;

    ffm_wmma_kernel<<<dim3(BATCH / BTILE), dim3(NTHR), 0, stream>>>(x, v, w1, w0, out);
}
